// ExpaModel_73478300500036
// MI455X (gfx1250) — compile-verified, matrix operand order checked by reading, not run on hardware
//
#include <hip/hip_runtime.h>
#include <hip/hip_bf16.h>
#include <stdint.h>

#define NODES 16384
#define EDGES 49152
#define DIM   768
#define HEADS 4
#define LAYERS 3
#define RELS  64
#define HD    (HEADS*DIM)     // 3072
#define ETOT  (EDGES+NODES)   // 65536
#define NEG_SLOPE 0.2f
#define LN_EPS 1e-5f

typedef __attribute__((ext_vector_type(16))) __bf16 v16bf;
typedef __attribute__((ext_vector_type(8)))  float  v8f;

union FragBF { v16bf v; unsigned int u[8]; };

__device__ __forceinline__ unsigned short f2bf(float f) {
  unsigned int u = __float_as_uint(f);
  u += 0x7FFFu + ((u >> 16) & 1u);          // round-to-nearest-even
  return (unsigned short)(u >> 16);
}
__device__ __forceinline__ float gelu_tanh(float x) {
  float x3 = x * x * x;
  return 0.5f * x * (1.0f + tanhf(0.7978845608028654f * (x + 0.044715f * x3)));
}
// order-preserving float<->uint encoding for atomicMax-based segment max
__device__ __forceinline__ unsigned int enc_f32(float f) {
  unsigned int b = __float_as_uint(f);
  return (b & 0x80000000u) ? ~b : (b | 0x80000000u);
}
__device__ __forceinline__ float dec_f32(unsigned int k) {
  unsigned int b = (k & 0x80000000u) ? (k & 0x7FFFFFFFu) : ~k;
  return __uint_as_float(b);
}

// ---------------- small prep kernels ----------------

__global__ void k_counts(const int* __restrict__ ei, const int* __restrict__ ea,
                         float* __restrict__ counts) {
  int e = blockIdx.x * blockDim.x + threadIdx.x;
  if (e < EDGES) {
    int dst = ei[EDGES + e];
    atomicAdd(&counts[(size_t)dst * RELS + ea[e]], 1.0f);
  }
}

__global__ void k_coef(const float* __restrict__ counts, unsigned short* __restrict__ coef_bf) {
  __shared__ float s[RELS];
  int n = blockIdx.x, r = threadIdx.x;
  float c = counts[(size_t)n * RELS + r];
  s[r] = c;
  __syncthreads();
  for (int off = RELS / 2; off > 0; off >>= 1) {
    if (r < off) s[r] += s[r + off];
    __syncthreads();
  }
  float deg = s[0] > 1.0f ? s[0] : 1.0f;
  coef_bf[(size_t)n * RELS + r] = f2bf(c / deg);
}

__global__ void k_f32_to_bf16(const float* __restrict__ in, unsigned short* __restrict__ out, int n) {
  int t = blockIdx.x * blockDim.x + threadIdx.x;
  if (t < n) out[t] = f2bf(in[t]);
}

// in: [rows, cols] f32 row-major -> out: [cols, rows] bf16 row-major (B^T for GEMM, K=rows)
__global__ void k_trans_f32_to_bf16(const float* __restrict__ in, unsigned short* __restrict__ out,
                                    int rows, int cols) {
  int t = blockIdx.x * blockDim.x + threadIdx.x;
  if (t < rows * cols) {
    int n = t % cols;
    int k = t / cols;
    out[(size_t)n * rows + k] = f2bf(in[t]);
  }
}

// ---------------- bf16 WMMA GEMM: C[M,Nc] = A[M,K] * B[K,Nc], Bt = B^T [Nc,K] ----------------
// grid = (Nc/64, M/16); block = 128 (4 waves); wave w computes tile [16 x 16] at col tileN+16w.
__global__ __launch_bounds__(128)
void k_gemm_bf16(const unsigned short* __restrict__ A, const unsigned short* __restrict__ Bt,
                 float* __restrict__ C, int M, int Nc, int K) {
  const int tileM = blockIdx.y * 16;
  const int tileN = blockIdx.x * 64 + (threadIdx.x >> 5) * 16;
  const int lane  = threadIdx.x & 31;
  const int half  = lane >> 4;       // K-group select per ISA 16-bit A layout
  const int r16   = lane & 15;

  const size_t arow = (size_t)(tileM + r16) * K;
  const size_t brow = (size_t)(tileN + r16) * K;

  v8f acc = {};
  for (int kk = 0; kk < K; kk += 32) {
    FragBF fa, fb;
#pragma unroll
    for (int i = 0; i < 8; ++i) {
      // lanes 0-15: K pairs {0,2,4,6,16,18,20,22}; lanes 16-31: +8
      int k = ((i < 4) ? 2 * i : 16 + 2 * (i - 4)) + half * 8;
      fa.u[i] = *(const unsigned int*)(A + arow + kk + k);
      fb.u[i] = *(const unsigned int*)(Bt + brow + kk + k);
    }
    acc = __builtin_amdgcn_wmma_f32_16x16x32_bf16(false, fa.v, false, fb.v,
                                                  (short)0, acc, false, false);
  }
#pragma unroll
  for (int r = 0; r < 8; ++r) {
    int row = tileM + r + half * 8;   // C/D layout: VGPR r -> M=r (lanes 0-15) / M=r+8
    C[(size_t)row * Nc + (tileN + r16)] = acc[r];
  }
}

// ---------------- edge phase ----------------

// one block (256 thr) per edge: logits[e,h] = att_h . leaky_relu(hs[src]+hd[dst]+he)
__global__ void k_edge_logits(const float* __restrict__ hs, const float* __restrict__ hd,
                              const float* __restrict__ heu, const float* __restrict__ heloop,
                              const int* __restrict__ ei, const int* __restrict__ ea,
                              const float* __restrict__ att_l,
                              float* __restrict__ logits, unsigned int* __restrict__ m_enc) {
  __shared__ float red[256];
  int e = blockIdx.x;
  int src, dst;
  const float* he_row;
  if (e < EDGES) {
    src = ei[e]; dst = ei[EDGES + e];
    he_row = heu + (size_t)ea[e] * HD;
  } else {
    src = e - EDGES; dst = src;
    he_row = heloop + (size_t)(e - EDGES) * HD;
  }
  const float* hs_row = hs + (size_t)src * HD;
  const float* hd_row = hd + (size_t)dst * HD;

  for (int h = 0; h < HEADS; ++h) {
    float p = 0.0f;
    for (int f = threadIdx.x; f < DIM; f += 256) {
      int idx = h * DIM + f;
      float z = hs_row[idx] + hd_row[idx] + he_row[idx];
      z = (z > 0.0f) ? z : NEG_SLOPE * z;
      p += z * att_l[idx];
    }
    red[threadIdx.x] = p;
    __syncthreads();
    for (int off = 128; off > 0; off >>= 1) {
      if (threadIdx.x < off) red[threadIdx.x] += red[threadIdx.x + off];
      __syncthreads();
    }
    if (threadIdx.x == 0) {
      float lg = red[0];
      logits[(size_t)e * HEADS + h] = lg;
      atomicMax(&m_enc[dst * HEADS + h], enc_f32(lg));
    }
    __syncthreads();
  }
}

// p = exp(logit - m[dst]); denom[dst] += p  (p overwrites logits buffer)
__global__ void k_edge_p(float* __restrict__ logits, const unsigned int* __restrict__ m_enc,
                         float* __restrict__ denom, const int* __restrict__ ei) {
  int t = blockIdx.x * blockDim.x + threadIdx.x;
  if (t < ETOT * HEADS) {
    int e = t >> 2, h = t & 3;
    int dst = (e < EDGES) ? ei[EDGES + e] : (e - EDGES);
    float m = dec_f32(m_enc[dst * HEADS + h]);
    float p = __expf(logits[t] - m);
    logits[t] = p;
    atomicAdd(&denom[dst * HEADS + h], p);
  }
}

// acc[dst,f] += (1/H) * sum_h alpha[e,h] * hs[src,h,f]
__global__ void k_edge_scatter(const float* __restrict__ hs, const float* __restrict__ p,
                               const float* __restrict__ denom, const int* __restrict__ ei,
                               float* __restrict__ acc) {
  int e = blockIdx.x;
  int src, dst;
  if (e < EDGES) { src = ei[e]; dst = ei[EDGES + e]; }
  else           { src = e - EDGES; dst = src; }
  float alpha[HEADS];
#pragma unroll
  for (int h = 0; h < HEADS; ++h)
    alpha[h] = p[(size_t)e * HEADS + h] / (denom[dst * HEADS + h] + 1e-16f) * (1.0f / HEADS);
  const float* hs_row = hs + (size_t)src * HD;
  for (int f = threadIdx.x; f < DIM; f += 256) {
    float v = alpha[0] * hs_row[f] + alpha[1] * hs_row[DIM + f]
            + alpha[2] * hs_row[2 * DIM + f] + alpha[3] * hs_row[3 * DIM + f];
    atomicAdd(&acc[(size_t)dst * DIM + f], v);
  }
}

// x_new = gelu(acc + bias) + x_old
__global__ void k_node_update(const float* __restrict__ acc, const float* __restrict__ bias_l,
                              const float* __restrict__ x_old, float* __restrict__ x_new) {
  int t = blockIdx.x * blockDim.x + threadIdx.x;
  if (t < NODES * DIM) {
    float v = acc[t] + bias_l[t % DIM];
    x_new[t] = gelu_tanh(v) + x_old[t];
  }
}

// out = gelu(LN(y + proj_b) * g + b), one block per row
__global__ void k_ln_gelu(const float* __restrict__ y, const float* __restrict__ pb,
                          const float* __restrict__ g, const float* __restrict__ b,
                          float* __restrict__ out) {
  __shared__ float s1[256], s2[256];
  int n = blockIdx.x;
  const float* row = y + (size_t)n * DIM;
  float v[3], a0 = 0.0f, a1 = 0.0f;
#pragma unroll
  for (int i = 0; i < 3; ++i) {
    int f = threadIdx.x + i * 256;
    float t = row[f] + pb[f];
    v[i] = t; a0 += t; a1 += t * t;
  }
  s1[threadIdx.x] = a0; s2[threadIdx.x] = a1;
  __syncthreads();
  for (int off = 128; off > 0; off >>= 1) {
    if (threadIdx.x < off) { s1[threadIdx.x] += s1[threadIdx.x + off]; s2[threadIdx.x] += s2[threadIdx.x + off]; }
    __syncthreads();
  }
  float mean = s1[0] * (1.0f / DIM);
  float var  = s2[0] * (1.0f / DIM) - mean * mean;
  float rstd = rsqrtf(var + LN_EPS);
#pragma unroll
  for (int i = 0; i < 3; ++i) {
    int f = threadIdx.x + i * 256;
    float t = (v[i] - mean) * rstd * g[f] + b[f];
    out[(size_t)n * DIM + f] = gelu_tanh(t);
  }
}

// ---------------- host orchestration ----------------

extern "C" void kernel_launch(void* const* d_in, const int* in_sizes, int n_in,
                              void* d_out, int out_size, void* d_ws, size_t ws_size,
                              hipStream_t stream) {
  const float* x      = (const float*)d_in[0];
  const float* rel    = (const float*)d_in[1];
  const float* w_src  = (const float*)d_in[2];
  const float* w_dst  = (const float*)d_in[3];
  const float* w_edge = (const float*)d_in[4];
  const float* att    = (const float*)d_in[5];
  const float* bias   = (const float*)d_in[6];
  const float* proj_w = (const float*)d_in[7];
  const float* proj_b = (const float*)d_in[8];
  const float* ln_g   = (const float*)d_in[9];
  const float* ln_b   = (const float*)d_in[10];
  const int*   eidx   = (const int*)d_in[11];
  const int*   eattr  = (const int*)d_in[12];
  float* out = (float*)d_out;
  (void)in_sizes; (void)n_in; (void)out_size; (void)ws_size;

  char* p = (char*)d_ws;
  auto alloc = [&](size_t bytes) -> char* {
    char* r = p;
    p += (bytes + 255) & ~(size_t)255;
    return r;
  };
  float* xa     = (float*)alloc(sizeof(float) * (size_t)NODES * DIM);
  float* xb     = (float*)alloc(sizeof(float) * (size_t)NODES * DIM);
  float* hs     = (float*)alloc(sizeof(float) * (size_t)NODES * HD);
  float* hd     = (float*)alloc(sizeof(float) * (size_t)NODES * HD);
  float* heloop = (float*)alloc(sizeof(float) * (size_t)NODES * HD);
  float* acc    = (float*)alloc(sizeof(float) * (size_t)NODES * DIM);   // also proj output
  float* counts = (float*)alloc(sizeof(float) * (size_t)NODES * RELS);
  float* heu    = (float*)alloc(sizeof(float) * (size_t)RELS * HD);
  float* logitsP= (float*)alloc(sizeof(float) * (size_t)ETOT * HEADS);
  unsigned int* m_enc = (unsigned int*)alloc(sizeof(unsigned int) * (size_t)NODES * HEADS);
  float* denom  = (float*)alloc(sizeof(float) * (size_t)NODES * HEADS);
  unsigned short* coef_bf = (unsigned short*)alloc(sizeof(unsigned short) * (size_t)NODES * RELS);
  unsigned short* x_bf    = (unsigned short*)alloc(sizeof(unsigned short) * (size_t)NODES * DIM);
  unsigned short* rel_bf  = (unsigned short*)alloc(sizeof(unsigned short) * (size_t)RELS * DIM);
  unsigned short* wT_bf   = (unsigned short*)alloc(sizeof(unsigned short) * (size_t)HD * DIM);
  unsigned short* heuT_bf = (unsigned short*)alloc(sizeof(unsigned short) * (size_t)HD * RELS);

  // self-loop attr factorization: counts -> coef
  hipMemsetAsync(counts, 0, sizeof(float) * (size_t)NODES * RELS, stream);
  k_counts<<<(EDGES + 255) / 256, 256, 0, stream>>>(eidx, eattr, counts);
  k_coef<<<NODES, RELS, 0, stream>>>(counts, coef_bf);
  k_f32_to_bf16<<<(RELS * DIM + 255) / 256, 256, 0, stream>>>(rel, rel_bf, RELS * DIM);

  const float* cur = x;
  float* nxt = xa;
  for (int l = 0; l < LAYERS; ++l) {
    k_f32_to_bf16<<<(NODES * DIM + 255) / 256, 256, 0, stream>>>(cur, x_bf, NODES * DIM);

    dim3 gW((DIM * HD + 255) / 256);
    // hs = x @ Wsrc
    k_trans_f32_to_bf16<<<gW, 256, 0, stream>>>(w_src + (size_t)l * DIM * HD, wT_bf, DIM, HD);
    k_gemm_bf16<<<dim3(HD / 64, NODES / 16), 128, 0, stream>>>(x_bf, wT_bf, hs, NODES, HD, DIM);
    // hd = x @ Wdst
    k_trans_f32_to_bf16<<<gW, 256, 0, stream>>>(w_dst + (size_t)l * DIM * HD, wT_bf, DIM, HD);
    k_gemm_bf16<<<dim3(HD / 64, NODES / 16), 128, 0, stream>>>(x_bf, wT_bf, hd, NODES, HD, DIM);
    // he_unique = rel_emb @ Wedge  (only 64 rows!)
    k_trans_f32_to_bf16<<<gW, 256, 0, stream>>>(w_edge + (size_t)l * DIM * HD, wT_bf, DIM, HD);
    k_gemm_bf16<<<dim3(HD / 64, RELS / 16), 128, 0, stream>>>(rel_bf, wT_bf, heu, RELS, HD, DIM);
    // he_loop = coef @ he_unique   (K = 64)
    k_trans_f32_to_bf16<<<(RELS * HD + 255) / 256, 256, 0, stream>>>(heu, heuT_bf, RELS, HD);
    k_gemm_bf16<<<dim3(HD / 64, NODES / 16), 128, 0, stream>>>(coef_bf, heuT_bf, heloop, NODES, HD, RELS);

    // edge softmax-attention phase
    hipMemsetAsync(m_enc, 0, sizeof(unsigned int) * (size_t)NODES * HEADS, stream); // enc(-inf)
    hipMemsetAsync(denom, 0, sizeof(float) * (size_t)NODES * HEADS, stream);
    hipMemsetAsync(acc, 0, sizeof(float) * (size_t)NODES * DIM, stream);
    k_edge_logits<<<ETOT, 256, 0, stream>>>(hs, hd, heu, heloop, eidx, eattr,
                                            att + (size_t)l * HD, logitsP, m_enc);
    k_edge_p<<<(ETOT * HEADS + 255) / 256, 256, 0, stream>>>(logitsP, m_enc, denom, eidx);
    k_edge_scatter<<<ETOT, 256, 0, stream>>>(hs, logitsP, denom, eidx, acc);
    k_node_update<<<(NODES * DIM + 255) / 256, 256, 0, stream>>>(acc, bias + (size_t)l * DIM, cur, nxt);

    cur = nxt;
    nxt = (cur == xa) ? xb : xa;
  }

  // projection -> LayerNorm -> GELU
  k_f32_to_bf16<<<(NODES * DIM + 255) / 256, 256, 0, stream>>>(cur, x_bf, NODES * DIM);
  k_trans_f32_to_bf16<<<(DIM * DIM + 255) / 256, 256, 0, stream>>>(proj_w, wT_bf, DIM, DIM);
  k_gemm_bf16<<<dim3(DIM / 64, NODES / 16), 128, 0, stream>>>(x_bf, wT_bf, acc, NODES, DIM, DIM);
  k_ln_gelu<<<NODES, 256, 0, stream>>>(acc, proj_b, ln_g, ln_b, out);
}
